// WSGATLayer_28235115003923
// MI455X (gfx1250) — matrix operand order repaired and checked by reading, not compiled, not run on hardware
//
#include <hip/hip_runtime.h>

#define NW 100000
#define NS 50000
#define NE 1000000
#define IN_DIM 256
#define OD 64

typedef __attribute__((ext_vector_type(16))) _Float16 v16h;
typedef __attribute__((ext_vector_type(8)))  float    v8f;

// ---- A fragment: 16x32 (MxK) tile from row-major f32, cvt to f16 ----------
// Layout (ISA 7.12.2): lane = 16*half + m ; VGPR j holds K = (j<4?0:16)+half*8+(j&3)*2 (+k0)
static __device__ __forceinline__ v16h load_a_frag(const float* __restrict__ rowBase,
                                                   int ld, int k0, int lane) {
  int m = lane & 15, half = lane >> 4;
  const float* r = rowBase + (size_t)m * ld + k0 + half * 8;
  v16h a;
#pragma unroll
  for (int j = 0; j < 8; ++j) {
    int k = ((j < 4) ? 0 : 16) + ((j & 3) << 1);
    a[2 * j]     = (_Float16)r[k];
    a[2 * j + 1] = (_Float16)r[k + 1];
  }
  return a;
}

// ---- Pre-packed B fragments (f16, fragment-ordered) -----------------------
// P holds frags in order frag = kstep*4 + ntile; each frag is 32 lanes x 16 halves,
// lane-contiguous so a fragment load is 32 contiguous bytes per lane (2x b128).
static __device__ __forceinline__ v16h load_b_packed(const _Float16* __restrict__ P,
                                                     int frag, int lane) {
  return *(const v16h*)(P + ((size_t)frag * 32 + lane) * 16);
}

// Build fragment-ordered f16 image of weight W [64][K] (B[k][n] = W[n][k]).
// grid = (K/32)*4 blocks of 32 threads; block b = kstep*4 + ntile.
__global__ void pack_w(const float* __restrict__ W, int K, _Float16* __restrict__ P) {
  int b = blockIdx.x, lane = threadIdx.x;
  int s = b >> 2, nt = b & 3;
  const float* r = W + (size_t)(nt * 16 + (lane & 15)) * K + s * 32 + ((lane >> 4) << 3);
  v16h f;
#pragma unroll
  for (int t = 0; t < 16; ++t) f[t] = (_Float16)r[t < 8 ? t : t + 8];
  *(v16h*)(P + ((size_t)b * 32 + lane) * 16) = f;
}

static __device__ __forceinline__ void atomicMaxFloat(float* addr, float v) {
  if (v >= 0.0f) atomicMax((int*)addr, __float_as_int(v));
  else           atomicMin((unsigned int*)addr, __float_as_uint(v));
}

// ---- K0: init scratch + output -------------------------------------------
__global__ void init_k(float* __restrict__ out, float* __restrict__ seg_max,
                       float* __restrict__ seg_sum) {
  int idx = blockIdx.x * blockDim.x + threadIdx.x;
  if (idx < NS * OD) out[idx] = 0.0f;
  if (idx < NS) {
    seg_max[idx] = __uint_as_float(0xff800000u); // -inf
    seg_sum[idx] = 0.0f;
  }
}

// ---- K1/K2: Z = X @ W.T  (X:[nrows,K] f32, W pre-packed f16) --------------
__global__ void proj_gemm(const float* __restrict__ X, const _Float16* __restrict__ P,
                          float* __restrict__ Z, int nrows, int K) {
  int gtid = blockIdx.x * blockDim.x + threadIdx.x;
  int wave = gtid >> 5, lane = gtid & 31;
  int ntiles = nrows >> 4;
  if (wave >= ntiles) return;
  const float* rowBase = X + (size_t)wave * 16 * K;
  v8f acc[4] = {};
  for (int k0 = 0; k0 < K; k0 += 32) {
    v16h a = load_a_frag(rowBase, K, k0, lane);
    int fb = (k0 >> 5) * 4;
#pragma unroll
    for (int nt = 0; nt < 4; ++nt) {
      v16h b = load_b_packed(P, fb + nt, lane);
      acc[nt] = __builtin_amdgcn_wmma_f32_16x16x32_f16(
          false, a, false, b, (short)0, acc[nt], false, false);
    }
  }
  int half = lane >> 4, nl = lane & 15;
#pragma unroll
  for (int nt = 0; nt < 4; ++nt)
#pragma unroll
    for (int j = 0; j < 8; ++j)
      Z[(size_t)(wave * 16 + j + 8 * half) * OD + nt * 16 + nl] = acc[nt][j];
}

// ---- K3: fused per-edge pass ---------------------------------------------
__global__ __launch_bounds__(128)
void edge_pass(const float* __restrict__ zw, const float* __restrict__ zsn,
               const float* __restrict__ tfidf,
               const _Float16* __restrict__ PfeatH, const _Float16* __restrict__ PgateH,
               const float* __restrict__ b_gate, const float* __restrict__ w_attn,
               const int* __restrict__ src, const int* __restrict__ dst,
               const int* __restrict__ root,
               float* __restrict__ e_buf, float* __restrict__ seg_max) {
  __shared__ float smem[4 * 3 * 1024]; // 4 waves x (zs|zd|z2)[16][64]
  int wib = threadIdx.x >> 5, lane = threadIdx.x & 31;
  int tile = blockIdx.x * 4 + wib;
  if (tile >= NE / 16) return;
  int tb = tile * 16;
  float* zsL = smem + wib * 3072;
  float* zdL = zsL + 1024;
  float* z2L = zdL + 1024;

  // gather: each lane owns half a row (32 floats) and moves it as 8x float4
  {
    int row = lane >> 1;              // 0..15
    int dbase = (lane & 1) * 32;      // 0 or 32
    int e = tb + row;
    int sI = src[e], dI = dst[e];
    int rI = root[dI];
    const float4* zsp = (const float4*)(zw  + (size_t)sI * OD + dbase);
    const float4* zdp = (const float4*)(zsn + (size_t)dI * OD + dbase);
    float* zsLr = zsL + row * 64 + dbase;
    float* zdLr = zdL + row * 64 + dbase;
    float* z2Lr = z2L + row * 64 + dbase;
#pragma unroll
    for (int q = 0; q < 8; ++q) {
      float4 a = zsp[q];
      float4 b = zdp[q];
      *(float4*)(zsLr + 4 * q) = a;
      *(float4*)(zdLr + 4 * q) = b;
      float4 c;
      c.x = rI ? a.x : b.x; c.y = rI ? a.y : b.y;
      c.z = rI ? a.z : b.z; c.w = rI ? a.w : b.w;
      *(float4*)(z2Lr + 4 * q) = c;
    }
  }

  // A fragments (wave-private LDS; compiler inserts dscnt waits)
  v16h aT0 = load_a_frag(tfidf + (size_t)tb * 64, 64, 0, lane);
  v16h aT1 = load_a_frag(tfidf + (size_t)tb * 64, 64, 32, lane);
  v16h aG0 = load_a_frag(z2L, 64, 0, lane);
  v16h aG1 = load_a_frag(z2L, 64, 32, lane);
  v16h aG2 = load_a_frag(zdL, 64, 0, lane);
  v16h aG3 = load_a_frag(zdL, 64, 32, lane);

  int half = lane >> 4, nl = lane & 15;
  float p[8];
#pragma unroll
  for (int j = 0; j < 8; ++j) p[j] = 0.0f;

#pragma unroll
  for (int nt = 0; nt < 4; ++nt) {
    int n = nt * 16 + nl;
    // dfeat fragment (2 WMMA, K=64)
    v8f df = {};
    df = __builtin_amdgcn_wmma_f32_16x16x32_f16(false, aT0, false,
        load_b_packed(PfeatH, nt, lane), (short)0, df, false, false);
    df = __builtin_amdgcn_wmma_f32_16x16x32_f16(false, aT1, false,
        load_b_packed(PfeatH, 4 + nt, lane), (short)0, df, false, false);
    // gate pre-activation (4 WMMA, K=128), C seeded with bias (column = lane&15)
    float bg = b_gate[n];
    v8f g;
#pragma unroll
    for (int j = 0; j < 8; ++j) g[j] = bg;
    g = __builtin_amdgcn_wmma_f32_16x16x32_f16(false, aG0, false,
        load_b_packed(PgateH, nt, lane), (short)0, g, false, false);
    g = __builtin_amdgcn_wmma_f32_16x16x32_f16(false, aG1, false,
        load_b_packed(PgateH, 4 + nt, lane), (short)0, g, false, false);
    g = __builtin_amdgcn_wmma_f32_16x16x32_f16(false, aG2, false,
        load_b_packed(PgateH, 8 + nt, lane), (short)0, g, false, false);
    g = __builtin_amdgcn_wmma_f32_16x16x32_f16(false, aG3, false,
        load_b_packed(PgateH, 12 + nt, lane), (short)0, g, false, false);

    float wa = w_attn[n];
#pragma unroll
    for (int j = 0; j < 8; ++j) {
      int m = j + 8 * half;
      float z2v = z2L[m * 64 + n];
      float zdv = zdL[m * 64 + n];
      float zsv = zsL[m * 64 + n];
      float gv  = 1.0f / (1.0f + __expf(-g[j]));
      float z22 = gv * tanhf(z2v) + (1.0f - gv) * zdv;
      float x   = z22 + zsv + df[j];
      float z3  = x > 0.0f ? x : 0.01f * x;
      p[j] += z3 * wa;
    }
  }

  // reduce over the 16 N-lanes of each half -> per-row logit e[m]
#pragma unroll
  for (int j = 0; j < 8; ++j) {
    float v = p[j];
    v += __shfl_xor(v, 1, 16);
    v += __shfl_xor(v, 2, 16);
    v += __shfl_xor(v, 4, 16);
    v += __shfl_xor(v, 8, 16);
    if (nl == 0) {
      int m = j + 8 * half;
      e_buf[tb + m] = v;
      atomicMaxFloat(&seg_max[dst[tb + m]], v);
    }
  }
}

// ---- K4: exp + segment sums via L2 float atomics -------------------------
__global__ void scatter_pass(const float* __restrict__ e_buf, const float* __restrict__ seg_max,
                             const float* __restrict__ zw,
                             const int* __restrict__ src, const int* __restrict__ dst,
                             float* __restrict__ seg_sum, float* __restrict__ out) {
  long long idx = (long long)blockIdx.x * blockDim.x + threadIdx.x;
  if (idx >= (long long)NE * OD) return;
  int e = (int)(idx >> 6), d = (int)(idx & 63);
  int ds = dst[e];
  float ex = __expf(e_buf[e] - seg_max[ds]);
  if (d == 0) atomicAdd(&seg_sum[ds], ex);
  float zsv = zw[(size_t)src[e] * OD + d];
  atomicAdd(&out[(size_t)ds * OD + d], ex * zsv);
}

// ---- K5: normalize -------------------------------------------------------
__global__ void normalize_k(float* __restrict__ out, const float* __restrict__ seg_sum) {
  int idx = blockIdx.x * blockDim.x + threadIdx.x;
  if (idx >= NS * OD) return;
  float den = seg_sum[idx >> 6];
  out[idx] = (den > 0.0f) ? out[idx] / den : 0.0f;
}

extern "C" void kernel_launch(void* const* d_in, const int* in_sizes, int n_in,
                              void* d_out, int out_size, void* d_ws, size_t ws_size,
                              hipStream_t stream) {
  const float* h      = (const float*)d_in[0];
  const float* o      = (const float*)d_in[1];
  const float* tfidf  = (const float*)d_in[2];
  const float* W_fc   = (const float*)d_in[3];
  const float* W_fc1  = (const float*)d_in[4];
  const float* W_feat = (const float*)d_in[5];
  const float* w_attn = (const float*)d_in[6];
  const float* W_gate = (const float*)d_in[7];
  const float* b_gate = (const float*)d_in[8];
  const int*   src    = (const int*)d_in[9];
  const int*   dst    = (const int*)d_in[10];
  const int*   root   = (const int*)d_in[11];
  float* out = (float*)d_out;

  // workspace layout (f32): z_w | z_s | e | seg_max | seg_sum | packed f16 weights
  float* zw      = (float*)d_ws;
  float* zs      = zw + (size_t)NW * OD;
  float* e_buf   = zs + (size_t)NS * OD;
  float* seg_max = e_buf + NE;
  float* seg_sum = seg_max + NS;
  _Float16* W_fcH   = (_Float16*)(seg_sum + NS);       // 8 ksteps * 4 nt * 512
  _Float16* W_fc1H  = W_fcH  + (size_t)32 * 512;       // 2 * 4 * 512
  _Float16* W_featH = W_fc1H + (size_t)8 * 512;        // 2 * 4 * 512
  _Float16* W_gateH = W_featH + (size_t)8 * 512;       // 4 * 4 * 512

  init_k<<<(NS * OD + 255) / 256, 256, 0, stream>>>(out, seg_max, seg_sum);
  pack_w<<<(IN_DIM / 32) * 4, 32, 0, stream>>>(W_fc,   IN_DIM, W_fcH);
  pack_w<<<(64 / 32) * 4,     32, 0, stream>>>(W_fc1,  64,     W_fc1H);
  pack_w<<<(64 / 32) * 4,     32, 0, stream>>>(W_feat, 64,     W_featH);
  pack_w<<<(128 / 32) * 4,    32, 0, stream>>>(W_gate, 128,    W_gateH);

  proj_gemm<<<((NW / 16) * 32 + 255) / 256, 256, 0, stream>>>(h, W_fcH,  zw, NW, IN_DIM);
  proj_gemm<<<((NS / 16) * 32 + 255) / 256, 256, 0, stream>>>(o, W_fc1H, zs, NS, 64);
  edge_pass<<<NE / 16 / 4, 128, 0, stream>>>(zw, zs, tfidf, W_featH, W_gateH,
                                             b_gate, w_attn, src, dst, root,
                                             e_buf, seg_max);
  long long tot = (long long)NE * OD;
  scatter_pass<<<(unsigned)((tot + 255) / 256), 256, 0, stream>>>(
      e_buf, seg_max, zw, src, dst, seg_sum, out);
  normalize_k<<<(NS * OD) / 256, 256, 0, stream>>>(out, seg_sum);
}
